// gru4recF_27118423507205
// MI455X (gfx1250) — matrix operand order repaired and checked by reading, not compiled, not run on hardware
//
#include <hip/hip_runtime.h>
#include <hip/hip_bf16.h>

// Problem constants (match reference)
#define BB    64
#define LL    200
#define V1    10001
#define EE    128
#define HH    128
#define BERT  768
#define BL    (BB * LL)     // 12800
#define G3    (3 * HH)      // 384

typedef __attribute__((ext_vector_type(16))) __bf16 v16bf;
typedef __attribute__((ext_vector_type(8)))  __bf16 v8bf;
typedef __attribute__((ext_vector_type(8)))  float  v8f;

// Native conversion (hardware RNE cvt, avoids integer-rounding VALU chains)
__device__ __forceinline__ __bf16 f2bf(float f) { return (__bf16)f; }

// WMMA lane-layout helpers (ISA 7.12.2, wave32, 16-bit operands)
// A 16x32 (code's k order, same for B; not the ISA table order): lane-half `half` holds 16 contiguous K at k0 = kb + half*16
// B 32x16: lane holds 16 contiguous K at (n = lane&15) row, k0 = kb + half*16
// C 16x16: element i -> M = i + half*8, N = lane&15
__device__ __forceinline__ int c_m(int i, int half) { return i + (half << 3); }

// Build an A fragment from a row pointer (one 32B chunk)
__device__ __forceinline__ v16bf load_a(const __bf16* row, int kb, int half) {
  return *(const v16bf*)(row + kb + half * 16);
}

// ---------------------------------------------------------------------------
// K0: convert fp32 weight matrices (and plot_emb) to bf16 in workspace
// ---------------------------------------------------------------------------
__global__ void __launch_bounds__(256)
k0_cvt(const float* proj_w, const float* w_ih, const float* w_hh, const float* out_w,
       const float* plot_emb,
       __bf16* proj_w_bf, __bf16* w_ih_bf, __bf16* w_hh_bf, __bf16* out_w_bf,
       __bf16* plot_bf) {
  const int tid = blockIdx.x * blockDim.x + threadIdx.x;
  const int stride = gridDim.x * blockDim.x;
  for (int t = tid; t < EE * BERT; t += stride) proj_w_bf[t] = f2bf(proj_w[t]);
  for (int t = tid; t < G3 * EE;   t += stride) w_ih_bf[t]  = f2bf(w_ih[t]);
  for (int t = tid; t < G3 * HH;   t += stride) w_hh_bf[t]  = f2bf(w_hh[t]);
  for (int t = tid; t < V1 * HH;   t += stride) out_w_bf[t] = f2bf(out_w[t]);
  for (int t = tid; t < V1 * BERT; t += stride) plot_bf[t]  = f2bf(plot_emb[t]);
}

// ---------------------------------------------------------------------------
// K1: e = movie_emb[x] + plot_emb[x] @ proj_w^T + proj_b   -> e_bf16 [BL,128]
// One wave computes a 16(M) x 128(N) tile; K = 768 in chunks of 32.
// ---------------------------------------------------------------------------
__global__ void __launch_bounds__(256)
k1_embed_proj(const int* x, const float* movie_emb, const __bf16* plot_bf,
              const __bf16* proj_w_bf, const float* proj_b, __bf16* e_bf) {
  const int lane = threadIdx.x & 31;
  const int wave = threadIdx.x >> 5;
  const int half = lane >> 4;
  const int l15  = lane & 15;
  const int wgid = blockIdx.x * (blockDim.x >> 5) + wave;  // 0..799
  const int mbase = wgid * 16;

  const int tokA = x[mbase + l15];
  const __bf16* arow = plot_bf + (long)tokA * BERT;

  v8f c[8];
#pragma unroll
  for (int t = 0; t < 8; ++t)
#pragma unroll
    for (int i = 0; i < 8; ++i) c[t][i] = 0.0f;

  for (int kb = 0; kb < BERT; kb += 32) {
    const v16bf a = load_a(arow, kb, half);
#pragma unroll
    for (int nt = 0; nt < 8; ++nt) {
      const v16bf b = *(const v16bf*)(proj_w_bf + (nt * 16 + l15) * BERT + kb + half * 16);
      c[nt] = __builtin_amdgcn_wmma_f32_16x16x32_bf16(false, a, false, b,
                                                      (short)0, c[nt], false, false);
    }
  }

  int tokc[8];
#pragma unroll
  for (int i = 0; i < 8; ++i) tokc[i] = x[mbase + c_m(i, half)];

#pragma unroll
  for (int nt = 0; nt < 8; ++nt) {
    const int n = nt * 16 + l15;
    const float bias = proj_b[n];
#pragma unroll
    for (int i = 0; i < 8; ++i) {
      const int row = mbase + c_m(i, half);
      float v = c[nt][i] + movie_emb[(long)tokc[i] * EE + n] + bias;
      e_bf[(long)row * EE + n] = f2bf(v);
    }
  }
}

// ---------------------------------------------------------------------------
// K2: gi = e @ w_ih^T + b_ih  -> fp32 [BL, 384]
// One wave computes 16(M) x 128(N); 3 waves cover N=384 per 16-row strip.
// ---------------------------------------------------------------------------
__global__ void __launch_bounds__(256)
k2_gi(const __bf16* e_bf, const __bf16* w_ih_bf, const float* b_ih, float* gi) {
  const int lane = threadIdx.x & 31;
  const int wave = threadIdx.x >> 5;
  const int half = lane >> 4;
  const int l15  = lane & 15;
  const int wgid = blockIdx.x * (blockDim.x >> 5) + wave;  // 0..2399
  const int strip = wgid / 3;
  const int part  = wgid % 3;
  const int mbase = strip * 16;
  const int nbase = part * 128;

  v8f c[8];
#pragma unroll
  for (int t = 0; t < 8; ++t)
#pragma unroll
    for (int i = 0; i < 8; ++i) c[t][i] = 0.0f;

  const __bf16* arow = e_bf + (long)(mbase + l15) * EE;
#pragma unroll
  for (int kb = 0; kb < EE; kb += 32) {
    const v16bf a = load_a(arow, kb, half);
#pragma unroll
    for (int nt = 0; nt < 8; ++nt) {
      const v16bf b = *(const v16bf*)(w_ih_bf + (nbase + nt * 16 + l15) * EE + kb + half * 16);
      c[nt] = __builtin_amdgcn_wmma_f32_16x16x32_bf16(false, a, false, b,
                                                      (short)0, c[nt], false, false);
    }
  }

#pragma unroll
  for (int nt = 0; nt < 8; ++nt) {
    const int n = nbase + nt * 16 + l15;
    const float bias = b_ih[n];
#pragma unroll
    for (int i = 0; i < 8; ++i) {
      const int row = mbase + c_m(i, half);
      gi[(long)row * G3 + n] = c[nt][i] + bias;
    }
  }
}

// ---------------------------------------------------------------------------
// K3: GRU recurrence. 4 blocks, each owns 16 batch rows. 256 threads = 8 waves.
// h kept twice in LDS: fp32 master (gate math) + bf16 shadow (WMMA A frags,
// zero per-step conversion work). gh staged in LDS fp32.
// ---------------------------------------------------------------------------
__global__ void __launch_bounds__(256)
k3_gru(const float* gi, const __bf16* w_hh_bf, const float* b_hh,
       const int* x_lens, const int* pack, __bf16* hs_bf) {
  __shared__ float  h_s[16][HH];
  __shared__ __bf16 hbf_s[16][HH];
  __shared__ float  gh_s[16][G3];

  const int lane = threadIdx.x & 31;
  const int wave = threadIdx.x >> 5;
  const int half = lane >> 4;
  const int l15  = lane & 15;
  const int bbase = blockIdx.x * 16;
  const int packv = pack[0];

  for (int f = threadIdx.x; f < 16 * HH; f += 256) {
    h_s[f >> 7][f & 127] = 0.0f;
    hbf_s[f >> 7][f & 127] = f2bf(0.0f);
  }
  __syncthreads();

  for (int t = 0; t < LL; ++t) {
    // ---- gh = h @ w_hh^T (A fragments straight from bf16 LDS shadow) ----
    v8f c[3];
#pragma unroll
    for (int s = 0; s < 3; ++s)
#pragma unroll
      for (int i = 0; i < 8; ++i) c[s][i] = 0.0f;

#pragma unroll
    for (int kb = 0; kb < HH; kb += 32) {
      const v16bf a = load_a(&hbf_s[l15][0], kb, half);
#pragma unroll
      for (int s = 0; s < 3; ++s) {
        const int nt = wave * 3 + s;
        const v16bf b = *(const v16bf*)(w_hh_bf + (nt * 16 + l15) * HH + kb + half * 16);
        c[s] = __builtin_amdgcn_wmma_f32_16x16x32_bf16(false, a, false, b,
                                                       (short)0, c[s], false, false);
      }
    }
#pragma unroll
    for (int s = 0; s < 3; ++s) {
      const int n = (wave * 3 + s) * 16 + l15;
#pragma unroll
      for (int i = 0; i < 8; ++i) gh_s[c_m(i, half)][n] = c[s][i];
    }
    __syncthreads();

    // ---- fused gates: r,z,n, h update, masked hs store ----
    for (int f = threadIdx.x; f < 16 * HH; f += 256) {
      const int r = f >> 7, j = f & 127;
      const int bidx = bbase + r;
      const long girow = ((long)bidx * LL + t) * G3;
      const float ir  = gi[girow + j];
      const float iz  = gi[girow + HH + j];
      const float in_ = gi[girow + 2 * HH + j];
      const float hr = gh_s[r][j] + b_hh[j];
      const float hz = gh_s[r][HH + j] + b_hh[HH + j];
      const float hn = gh_s[r][2 * HH + j] + b_hh[2 * HH + j];
      const float rg = 1.0f / (1.0f + __expf(-(ir + hr)));
      const float zg = 1.0f / (1.0f + __expf(-(iz + hz)));
      const float ng = tanhf(in_ + rg * hn);
      const float hold = h_s[r][j];
      const float hnew = (1.0f - zg) * ng + zg * hold;
      h_s[r][j] = hnew;
      hbf_s[r][j] = f2bf(hnew);
      const bool valid = (packv == 0) || (t < x_lens[bidx]);
      hs_bf[((long)bidx * LL + t) * HH + j] = f2bf(valid ? hnew : 0.0f);
    }
    __syncthreads();
  }
}

// ---------------------------------------------------------------------------
// K4: logits = hs @ out_w^T + out_b  -> fp32 [BL, 10001]   (store-bound, 512MB)
// Block = 8 waves sharing one 128-col N-group; out_w tile (128x128 bf16, 32KB)
// staged in LDS once and reused by all 8 M-strips. Loads clamped, stores
// guarded, so EXEC is all-1s around every WMMA.
// ---------------------------------------------------------------------------
__global__ void __launch_bounds__(256)
k4_logits(const __bf16* hs_bf, const __bf16* out_w_bf, const float* out_b, float* out) {
  __shared__ __bf16 lds_b[128][HH];   // [n_local][k], 32KB

  const int lane = threadIdx.x & 31;
  const int wave = threadIdx.x >> 5;
  const int half = lane >> 4;
  const int l15  = lane & 15;
  const int ng    = blockIdx.x % 79;        // N-group: 79 * 128 >= 10001
  const int rowg  = blockIdx.x / 79;        // 0..99 row-groups of 128 rows
  const int nbase = ng * 128;
  const int mbase = rowg * 128 + wave * 16; // this wave's 16-row strip

  // ---- stage out_w tile into LDS (rows clamped at V1-1; never stored) ----
  {
    const int rr  = threadIdx.x >> 1;        // 0..127 local col
    const int seg = (threadIdx.x & 1) * 64;  // element offset (128B halves)
    const int n  = nbase + rr;
    const int nc = (n < V1) ? n : (V1 - 1);
    const v8bf* src = (const v8bf*)(out_w_bf + (long)nc * HH + seg);
    v8bf* dst = (v8bf*)(&lds_b[rr][seg]);
#pragma unroll
    for (int q = 0; q < 8; ++q) dst[q] = src[q];
  }
  __syncthreads();

  v8f c[8];
#pragma unroll
  for (int t = 0; t < 8; ++t)
#pragma unroll
    for (int i = 0; i < 8; ++i) c[t][i] = 0.0f;

  const __bf16* arow = hs_bf + (long)(mbase + l15) * HH;
#pragma unroll
  for (int kb = 0; kb < HH; kb += 32) {
    const v16bf a = load_a(arow, kb, half);
#pragma unroll
    for (int nt = 0; nt < 8; ++nt) {
      const v16bf b = *(const v16bf*)(&lds_b[nt * 16 + l15][kb + half * 16]);
      c[nt] = __builtin_amdgcn_wmma_f32_16x16x32_bf16(false, a, false, b,
                                                      (short)0, c[nt], false, false);
    }
  }

#pragma unroll
  for (int nt = 0; nt < 8; ++nt) {
    const int n = nbase + nt * 16 + l15;
    if (n < V1) {
      const float bias = out_b[n];
#pragma unroll
      for (int i = 0; i < 8; ++i) {
        const int row = mbase + c_m(i, half);
        out[(long)row * V1 + n] = c[nt][i] + bias;
      }
    }
  }
}

// ---------------------------------------------------------------------------
extern "C" void kernel_launch(void* const* d_in, const int* in_sizes, int n_in,
                              void* d_out, int out_size, void* d_ws, size_t ws_size,
                              hipStream_t stream) {
  (void)in_sizes; (void)n_in; (void)out_size; (void)ws_size;

  const int*   x        = (const int*)d_in[0];
  const int*   x_lens   = (const int*)d_in[1];
  const float* movie_emb= (const float*)d_in[2];
  const float* plot_emb = (const float*)d_in[3];
  const float* proj_w   = (const float*)d_in[4];
  const float* proj_b   = (const float*)d_in[5];
  const float* w_ih     = (const float*)d_in[6];
  const float* w_hh     = (const float*)d_in[7];
  const float* b_ih     = (const float*)d_in[8];
  const float* b_hh     = (const float*)d_in[9];
  const float* out_w    = (const float*)d_in[10];
  const float* out_b    = (const float*)d_in[11];
  const int*   pack     = (const int*)d_in[12];
  float*       out      = (float*)d_out;

  // Workspace carve-up (256B aligned)
  char* ws = (char*)d_ws;
  size_t off = 0;
  auto carve = [&](size_t bytes) {
    void* p = ws + off;
    off += (bytes + 255) & ~(size_t)255;
    return p;
  };
  __bf16* proj_w_bf = (__bf16*)carve((size_t)EE * BERT * 2);
  __bf16* w_ih_bf   = (__bf16*)carve((size_t)G3 * EE * 2);
  __bf16* w_hh_bf   = (__bf16*)carve((size_t)G3 * HH * 2);
  __bf16* out_w_bf  = (__bf16*)carve((size_t)V1 * HH * 2);
  __bf16* plot_bf   = (__bf16*)carve((size_t)V1 * BERT * 2);
  __bf16* e_bf      = (__bf16*)carve((size_t)BL * EE * 2);
  float*  gi        = (float*) carve((size_t)BL * G3 * 4);
  __bf16* hs_bf     = (__bf16*)carve((size_t)BL * HH * 2);

  // K0: weight + plot_emb conversion (grid-stride; ~37MB read, one-shot)
  k0_cvt<<<1024, 256, 0, stream>>>(proj_w, w_ih, w_hh, out_w, plot_emb,
                                   proj_w_bf, w_ih_bf, w_hh_bf, out_w_bf, plot_bf);

  // K1: 800 wave-tiles (16 rows each), 8 waves/block -> 100 blocks
  k1_embed_proj<<<100, 256, 0, stream>>>(x, movie_emb, plot_bf,
                                         proj_w_bf, proj_b, e_bf);

  // K2: 800 strips * 3 N-parts = 2400 waves -> 300 blocks
  k2_gi<<<300, 256, 0, stream>>>(e_bf, w_ih_bf, b_ih, gi);

  // K3: 4 blocks * 16 batch rows, sequential over L=200 inside
  k3_gru<<<4, 256, 0, stream>>>(gi, w_hh_bf, b_hh, x_lens, pack, hs_bf);

  // K4: 100 row-groups (128 rows) * 79 N-groups = 7900 blocks
  k4_logits<<<7900, 256, 0, stream>>>(hs_bf, out_w_bf, out_b, out);
}
